// SelfAttention_5669356830734
// MI455X (gfx1250) — matrix operand order repaired and checked by reading, not compiled, not run on hardware
//
#include <hip/hip_runtime.h>

#define SEQ 4096
#define EMB 2048

typedef unsigned short u16;
typedef __attribute__((ext_vector_type(16))) __bf16 bf16x16;
typedef __attribute__((ext_vector_type(8)))  float  f32x8;

union FragU { bf16x16 v; uint4 q[2]; };

__device__ __forceinline__ u16 f2bf(float x) {
    unsigned u = __builtin_bit_cast(unsigned int, x);
    u += 0x7FFFu + ((u >> 16) & 1u);          // round-to-nearest-even
    return (u16)(u >> 16);
}

// ---- gfx1250 async global->LDS path (no VGPR data movement, ASYNCcnt) ----
#if defined(__HIP_DEVICE_COMPILE__) && __has_builtin(__builtin_amdgcn_global_load_async_to_lds_b128)
#define HAVE_ASYNC_LDS 1
#else
#define HAVE_ASYNC_LDS 0
#endif

#if HAVE_ASYNC_LDS
typedef int v4i_vs __attribute__((vector_size(16)));
typedef __attribute__((address_space(1))) v4i_vs* gptr128_t;
typedef __attribute__((address_space(3))) v4i_vs* lptr128_t;

__device__ __forceinline__ gptr128_t as_global(const void* p) {
    return (gptr128_t)(unsigned long long)p;
}
__device__ __forceinline__ lptr128_t as_local(void* p) {
    // LDS aperture: generic addr low 32 bits == LDS offset (ISA §10.2)
    return (lptr128_t)(unsigned int)(unsigned long long)p;
}
#if __has_builtin(__builtin_amdgcn_s_wait_asynccnt)
#define WAIT_ASYNC(n) __builtin_amdgcn_s_wait_asynccnt(n)
#else
#define WAIT_ASYNC(n) asm volatile("s_wait_asynccnt %0" ::"n"(n) : "memory")
#endif
#else
#define WAIT_ASYNC(n)
#endif

// ---------------------------------------------------------------------------
// f32 -> bf16 bulk convert
// ---------------------------------------------------------------------------
__global__ __launch_bounds__(256) void conv_f32_bf16(const float* __restrict__ in,
                                                     u16* __restrict__ out, int n) {
    int idx = blockIdx.x * blockDim.x + threadIdx.x;
    int stride = gridDim.x * blockDim.x;
    for (int i = idx; i < n; i += stride) out[i] = f2bf(in[i]);
}

// ---------------------------------------------------------------------------
// LDS-staged NT-GEMM via v_wmma_f32_16x16x32_bf16:
//   C[m,n] = sum_k A[m,k] * B[n,k]   (both operands k-contiguous, bf16)
//
// Block = 256 threads = 8 waves (2 M-rows x 4 N-cols). Block tile
// 128(M) x 256(N) x 32(K), double-buffered LDS (60 KB), filled by
// global_load_async_to_lds_b128 (ASYNCcnt; zero VGPR data movement).
// Per-wave tile 64 x 64: acc[4][4], A/B-frag reuse x4 => ~87 flop/L2-byte,
// 16 WMMAs per barrier pair. LDS rows padded to 40 u16 (80 B) for
// conflict-free ds_load_b128.
//
// OUTK: 0 = bf16 row-major, 1 = bf16 transposed, 2 = f32 row-major
// CSKIP: skip whole blocks above the causal diagonal (score GEMM)
// CKLIM: clamp k-loop to block-uniform causal limit (P·V GEMM)
// ---------------------------------------------------------------------------
template <int OUTK, bool CSKIP, bool CKLIM>
__global__ __launch_bounds__(256) void gemm_nt_bf16(const u16* __restrict__ A,
                                                    const u16* __restrict__ B,
                                                    void* __restrict__ C,
                                                    int M, int N, int K,
                                                    int lda, int ldb, int ldc) {
    constexpr int BM = 128, BN = 256, BK = 32, LSTR = 40;  // LSTR: padded row (u16)
    __shared__ u16 As[2][BM * LSTR];   // 2 x 10 KB
    __shared__ u16 Bs[2][BN * LSTR];   // 2 x 20 KB

    const int tid  = threadIdx.x;
    const int lane = tid & 31;
    const int wave = tid >> 5;
    const int blockM = blockIdx.y * BM;
    const int blockN = blockIdx.x * BN;

    if (CSKIP && blockN > blockM + (BM - 1)) return;
    const int kEnd = CKLIM ? ((blockM + BM < K) ? (blockM + BM) : K) : K;

    const int wm = (wave >> 2) * 64;   // wave row offset within block (0/64)
    const int wn = (wave & 3) * 64;    // wave col offset within block (0..192)

    f32x8 acc[4][4];
    const f32x8 fzero = {0.f, 0.f, 0.f, 0.f, 0.f, 0.f, 0.f, 0.f};
#pragma unroll
    for (int i = 0; i < 4; ++i)
#pragma unroll
        for (int j = 0; j < 4; ++j) acc[i][j] = fzero;

    // ---- staging map: rows (tid>>2)+64*i, 16B column (tid&3).
    //      A: 128 rows (i=0..1), B: 256 rows (i=0..3); 16B vectors.
    const int srow = tid >> 2;          // 0..63
    const int scol = (tid & 3) * 8;     // 0,8,16,24
    const u16* gA = A + (size_t)(blockM + srow) * lda + scol;
    const u16* gB = B + (size_t)(blockN + srow) * ldb + scol;
    const int soff = srow * LSTR + scol;

    auto fillTile = [&](int buf, int k) {
#if HAVE_ASYNC_LDS
#pragma unroll
        for (int i = 0; i < 2; ++i)
            __builtin_amdgcn_global_load_async_to_lds_b128(
                as_global(gA + (size_t)(64 * i) * lda + k),
                as_local(&As[buf][soff + 64 * i * LSTR]), 0, 0);
#pragma unroll
        for (int i = 0; i < 4; ++i)
            __builtin_amdgcn_global_load_async_to_lds_b128(
                as_global(gB + (size_t)(64 * i) * ldb + k),
                as_local(&Bs[buf][soff + 64 * i * LSTR]), 0, 0);
#else
#pragma unroll
        for (int i = 0; i < 2; ++i) {
            uint4 ta = *(const uint4*)(gA + (size_t)(64 * i) * lda + k);
            *(uint4*)(&As[buf][soff + 64 * i * LSTR]) = ta;
        }
#pragma unroll
        for (int i = 0; i < 4; ++i) {
            uint4 tb = *(const uint4*)(gB + (size_t)(64 * i) * ldb + k);
            *(uint4*)(&Bs[buf][soff + 64 * i * LSTR]) = tb;
        }
#endif
    };

    // ---- per-lane fragment coordinates (ISA 16-bit A 16x32 / B 32x16 layouts)
    const int arow   = wm + (lane & 15);
    const int khalfA = (lane >> 4) << 3;   // 0 or 8
    const int bn     = lane & 15;
    const int koffB  = (lane >> 4) << 3;   // 0 or 8

    auto computeTile = [&](int pb) {
        FragU aF[4], bF[4];
#pragma unroll
        for (int i = 0; i < 4; ++i) {
            const int base = (arow + 16 * i) * LSTR + khalfA;
            aF[i].q[0] = *(const uint4*)(&As[pb][base]);
            aF[i].q[1] = *(const uint4*)(&As[pb][base + 16]);
        }
#pragma unroll
        for (int j = 0; j < 4; ++j) {
            const int base = (wn + 16 * j + bn) * LSTR + koffB;
            bF[j].q[0] = *(const uint4*)(&Bs[pb][base]);
            bF[j].q[1] = *(const uint4*)(&Bs[pb][base + 16]);
        }
#pragma unroll
        for (int i = 0; i < 4; ++i)
#pragma unroll
            for (int j = 0; j < 4; ++j)
                acc[i][j] = __builtin_amdgcn_wmma_f32_16x16x32_bf16(
                    false, aF[i].v, false, bF[j].v, (short)0, acc[i][j], false, false);
    };

    fillTile(0, 0);
    int p = 0;
    int k = 0;
    // steady state: every iteration has a successor tile
    for (; k + BK < kEnd; k += BK) {
        fillTile(p ^ 1, k + BK);
        // async loads retire in order: waiting <=6 guarantees this wave's 6
        // older (current-buffer) transfers have landed in LDS.
        WAIT_ASYNC(6);
        __syncthreads();
        computeTile(p);
        __syncthreads();
        p ^= 1;
    }
    // peeled last iteration
    WAIT_ASYNC(0);
    __syncthreads();
    computeTile(p);

    // D fragment layout: VGPR r, lane -> M = r + 8*(lane>>4), N = lane&15.
    const int mrow = (lane >> 4) << 3;
#pragma unroll
    for (int i = 0; i < 4; ++i)
#pragma unroll
        for (int j = 0; j < 4; ++j)
#pragma unroll
            for (int r = 0; r < 8; ++r) {
                const int m = blockM + wm + 16 * i + mrow + r;
                const int n = blockN + wn + 16 * j + bn;
                const float v = acc[i][j][r];
                if constexpr (OUTK == 2)
                    ((float*)C)[(size_t)m * ldc + n] = v;
                else if constexpr (OUTK == 0)
                    ((u16*)C)[(size_t)m * ldc + n] = f2bf(v);
                else
                    ((u16*)C)[(size_t)n * ldc + m] = f2bf(v);   // transposed (Vt)
            }
}

// ---------------------------------------------------------------------------
// Causal row softmax over the f32 score matrix; writes bf16 P in-place
// (P row i at byte offset i*16384, stride 2*SEQ u16; entries j>i written 0).
// ---------------------------------------------------------------------------
__global__ __launch_bounds__(256) void softmax_causal(float* __restrict__ S) {
    __shared__ float rowbuf[SEQ];    // 16 KB
    __shared__ float red[256];
    const int i = blockIdx.x;
    const int tid = threadIdx.x;
    const float scale = 0.022097086912079608f;   // 1/sqrt(2048)
    const float NEG = -3.0e38f;

    const float* srow = S + (size_t)i * SEQ;
    float m = NEG;
    for (int j = tid; j < SEQ; j += 256) {
        float v = (j <= i) ? srow[j] * scale : NEG;
        rowbuf[j] = v;
        m = fmaxf(m, v);
    }
    red[tid] = m; __syncthreads();
    for (int s = 128; s > 0; s >>= 1) {
        if (tid < s) red[tid] = fmaxf(red[tid], red[tid + s]);
        __syncthreads();
    }
    m = red[0]; __syncthreads();

    float l = 0.f;
    for (int j = tid; j < SEQ; j += 256) {
        float e = (j <= i) ? __expf(rowbuf[j] - m) : 0.f;
        rowbuf[j] = e;
        l += e;
    }
    red[tid] = l; __syncthreads();
    for (int s = 128; s > 0; s >>= 1) {
        if (tid < s) red[tid] += red[tid + s];
        __syncthreads();
    }
    const float inv = 1.0f / red[0];

    u16* prow = (u16*)S + (size_t)i * (2 * SEQ);   // overwrite dead score row
    for (int j = tid; j < SEQ; j += 256) prow[j] = f2bf(rowbuf[j] * inv);
}

// ---------------------------------------------------------------------------
extern "C" void kernel_launch(void* const* d_in, const int* in_sizes, int n_in,
                              void* d_out, int out_size, void* d_ws, size_t ws_size,
                              hipStream_t stream) {
    (void)in_sizes; (void)n_in; (void)out_size; (void)ws_size;
    const float* X  = (const float*)d_in[0];
    const float* Wq = (const float*)d_in[1];
    const float* Wk = (const float*)d_in[2];
    const float* Wv = (const float*)d_in[3];
    float* Out = (float*)d_out;

    // Workspace layout (~152 MB total)
    char* p = (char*)d_ws;
    u16* Xbf = (u16*)p;  p += (size_t)SEQ * EMB * sizeof(u16);
    u16* Wqb = (u16*)p;  p += (size_t)EMB * EMB * sizeof(u16);
    u16* Wkb = (u16*)p;  p += (size_t)EMB * EMB * sizeof(u16);
    u16* Wvb = (u16*)p;  p += (size_t)EMB * EMB * sizeof(u16);
    u16* Qbf = (u16*)p;  p += (size_t)SEQ * EMB * sizeof(u16);
    u16* Kbf = (u16*)p;  p += (size_t)SEQ * EMB * sizeof(u16);
    u16* Vt  = (u16*)p;  p += (size_t)EMB * SEQ * sizeof(u16);  // [EMB][SEQ]
    float* Sf = (float*)p;                                      // [SEQ][SEQ] f32 / P bf16

    // 1) convert operands to bf16
    conv_f32_bf16<<<4096, 256, 0, stream>>>(X,  Xbf, SEQ * EMB);
    conv_f32_bf16<<<2048, 256, 0, stream>>>(Wq, Wqb, EMB * EMB);
    conv_f32_bf16<<<2048, 256, 0, stream>>>(Wk, Wkb, EMB * EMB);
    conv_f32_bf16<<<2048, 256, 0, stream>>>(Wv, Wvb, EMB * EMB);

    const dim3 blk(256);
    // 2) Q = X·Wqᵀ, K = X·Wkᵀ  (bf16 row-major [SEQ][EMB])
    gemm_nt_bf16<0, false, false><<<dim3(EMB / 256, SEQ / 128), blk, 0, stream>>>(
        Xbf, Wqb, Qbf, SEQ, EMB, EMB, EMB, EMB, EMB);
    gemm_nt_bf16<0, false, false><<<dim3(EMB / 256, SEQ / 128), blk, 0, stream>>>(
        Xbf, Wkb, Kbf, SEQ, EMB, EMB, EMB, EMB, EMB);
    // 3) Vt = (X·Wvᵀ)ᵀ  (bf16 [EMB][SEQ] so P·V contracts contiguously)
    gemm_nt_bf16<1, false, false><<<dim3(EMB / 256, SEQ / 128), blk, 0, stream>>>(
        Xbf, Wvb, Vt, SEQ, EMB, EMB, EMB, EMB, /*ldc=*/SEQ);
    // 4) S = Q·Kᵀ  (f32, causal blocks only)
    gemm_nt_bf16<2, true, false><<<dim3(SEQ / 256, SEQ / 128), blk, 0, stream>>>(
        Qbf, Kbf, Sf, SEQ, SEQ, EMB, EMB, EMB, SEQ);
    // 5) causal softmax -> bf16 P in-place over S
    softmax_causal<<<SEQ, 256, 0, stream>>>(Sf);
    // 6) Out = P·V  (A = P with lda = 2*SEQ u16, k-loop causally clamped)
    gemm_nt_bf16<2, false, true><<<dim3(EMB / 256, SEQ / 128), blk, 0, stream>>>(
        (const u16*)Sf, Vt, Out, SEQ, EMB, SEQ, /*lda=*/2 * SEQ, /*ldb=*/SEQ, /*ldc=*/EMB);
}
